// WindowAttention_22290880266855
// MI455X (gfx1250) — compile-verified, matrix operand order checked by reading, not run on hardware
//
#include <hip/hip_runtime.h>
#include <hip/hip_bf16.h>
#include <math.h>

typedef __attribute__((ext_vector_type(16))) _Float16 v16h;
typedef __attribute__((ext_vector_type(8)))  _Float16 v8h;
typedef __attribute__((ext_vector_type(8)))  float    v8f;

#define WMMA_F16(a, b, c) \
    __builtin_amdgcn_wmma_f32_16x16x32_f16(false, (a), false, (b), (short)0, (c), false, false)

// ---------------------------------------------------------------------------
// Weight fragment packing: B-matrix register layout for v_wmma_f32_16x16x32_f16.
// Fragment = one (16-col output tile) x (K=32 step). Lane l of the wave holds
// 16 halves: n = ntile*16 + (l&15); kb = kstep*32 + ((l>>4)<<3);
// halves 0..7 = W[n][kb+0..7], halves 8..15 = W[n][kb+16..23].
// Packed buffer: pk[((frag*32)+lane)*16 + j]  -> 32 contiguous bytes per lane.
// ---------------------------------------------------------------------------
constexpr int QKV_NTILES  = 48;              // 768 / 16
constexpr int PROJ_NTILES = 16;              // 256 / 16
constexpr int NKSTEPS     = 8;               // 256 / 32
constexpr int QKV_FRAGS   = QKV_NTILES * NKSTEPS;    // 384
constexpr int PROJ_FRAGS  = PROJ_NTILES * NKSTEPS;   // 128
constexpr int TOTAL_FRAGS = QKV_FRAGS + PROJ_FRAGS;  // 512 -> 512*32*16*2 B = 512 KB

__global__ __launch_bounds__(256)
void pack_weights_kernel(const float* __restrict__ qkv_w,
                         const float* __restrict__ proj_w,
                         _Float16* __restrict__ pk) {
    int tid  = blockIdx.x * 256 + threadIdx.x;
    int frag = tid >> 5, lane = tid & 31;
    if (frag >= TOTAL_FRAGS) return;
    const float* W;
    int f;
    if (frag < QKV_FRAGS) { W = qkv_w; f = frag; }
    else                  { W = proj_w; f = frag - QKV_FRAGS; }
    int ntile = f >> 3, ks = f & 7;
    int n  = ntile * 16 + (lane & 15);
    int kb = ks * 32 + ((lane >> 4) << 3);
    const float* p = W + (size_t)n * 256 + kb;
    _Float16* d = pk + ((size_t)frag * 32 + lane) * 16;
#pragma unroll
    for (int j = 0; j < 8; ++j) {
        d[j]     = (_Float16)p[j];
        d[8 + j] = (_Float16)p[16 + j];
    }
}

// Pre-packed B fragment: two contiguous 16B loads per lane (global_load_b128).
__device__ __forceinline__ v16h pkFrag(const _Float16* __restrict__ pk, int frag, int lane) {
    const v8h* p = (const v8h*)(pk + ((size_t)frag * 32 + lane) * 16);
    v8h lo = p[0];
    v8h hi = p[1];
    v16h b;
#pragma unroll
    for (int j = 0; j < 8; ++j) { b[j] = lo[j]; b[8 + j] = hi[j]; }
    return b;
}

// A-fragment (16 rows x K=32) from row-major f16 LDS (also valid as B when the
// source is stored [n][k]).  Two ds_load_b128 per lane.
__device__ __forceinline__ v16h ldsFragRowK(const _Float16* p, int row0, int k0,
                                            int stride, int lane) {
    int r  = row0 + (lane & 15);
    int kb = k0 + ((lane >> 4) << 3);
    const v8h* q = (const v8h*)(p + r * stride + kb);
    v8h lo = q[0];
    v8h hi = q[2];   // +16 halves
    v16h a;
#pragma unroll
    for (int j = 0; j < 8; ++j) { a[j] = lo[j]; a[8 + j] = hi[j]; }
    return a;
}

__global__ __launch_bounds__(256, 1)
void winattn_fused_kernel(const float* __restrict__ x,
                          const _Float16* __restrict__ wpk,   // packed f16 weights
                          const float* __restrict__ qkv_b,
                          const float* __restrict__ bias_table,
                          const float* __restrict__ proj_b,
                          float* __restrict__ out) {
    constexpr int N = 66, NP = 80, C = 256, H = 8, HD = 32, APW = 96;
    const int win  = blockIdx.x;
    const int tid  = threadIdx.x;
    const int lane = tid & 31;
    const int wave = tid >> 5;

    __shared__ _Float16 lds_x [NP * C];    // 40960 B  x window, f16, zero-padded rows
    __shared__ _Float16 lds_q [NP * HD];   //  5120 B  per-head Q (pre-scaled, +bias)
    __shared__ _Float16 lds_k [NP * HD];   //  5120 B
    __shared__ _Float16 lds_vt[HD * APW];  //  6144 B  V^T [hd][token], token-padded to 96
    __shared__ float    lds_af[NP * NP];   // 25600 B  raw scores + rel-pos bias (f32)
    __shared__ _Float16 lds_ah[NP * APW];  // 15360 B  softmaxed attn f16, K-padded to 96
    __shared__ _Float16 lds_o [NP * C];    // 40960 B  pre-projection output
    __shared__ float    row_mx[NP], row_inv[NP];

    // ---- Phase A: stage x -> LDS f16; zero pads; zero V^T pad columns once ----
    const float* xw = x + (size_t)win * N * C;
    for (int i = tid; i < N * C; i += 256)          lds_x[i] = (_Float16)xw[i];
    for (int i = N * C + tid; i < NP * C; i += 256) lds_x[i] = (_Float16)0.f;
    for (int i = tid; i < HD * APW; i += 256)       lds_vt[i] = (_Float16)0.f;
    __syncthreads();

    const float scale = 0.17677669529663687f;  // 1/sqrt(hd=32)

    for (int h = 0; h < H; ++h) {
        // ---- Phase B: QKV for this head (30 tiles: 3 mats x 5 M x 2 N) ----
        for (int t = wave; t < 30; t += 8) {
            int mat = t / 10, rem = t % 10, mt = rem >> 1, nt = rem & 1;
            int gn      = mat * C + h * HD + nt * 16;          // qkv_b index base
            int ntileG  = mat * 16 + h * 2 + nt;               // global 16-col tile of W
            v8f c = {};
#pragma unroll
            for (int ks = 0; ks < NKSTEPS; ++ks) {
                v16h a = ldsFragRowK(lds_x, mt * 16, ks * 32, C, lane);
                v16h b = pkFrag(wpk, ntileG * NKSTEPS + ks, lane);
                c = WMMA_F16(a, b, c);
            }
            int n = lane & 15, mh = (lane >> 4) << 3;
            float bb = qkv_b[gn + n];
            if (mat < 2) {
                _Float16* dst = (mat == 0) ? lds_q : lds_k;
#pragma unroll
                for (int r = 0; r < 8; ++r) {
                    int m = mt * 16 + mh + r;
                    float v = (m < N) ? (c[r] + bb) : 0.f;     // pad rows exactly zero
                    if (mat == 0) v *= scale;                  // fold q*scale
                    dst[m * HD + nt * 16 + n] = (_Float16)v;
                }
            } else {
                // store V transposed: vt[hd_col][token]
#pragma unroll
                for (int r = 0; r < 8; ++r) {
                    int m = mt * 16 + mh + r;
                    float v = (m < N) ? (c[r] + bb) : 0.f;
                    lds_vt[(nt * 16 + n) * APW + m] = (_Float16)v;
                }
            }
        }
        __syncthreads();

        // ---- Phase C: scores = q_s @ k^T (+rel-pos bias), 25 tiles, K=32 ----
        for (int t = wave; t < 25; t += 8) {
            int mt = t / 5, nt = t % 5;
            v16h a = ldsFragRowK(lds_q, mt * 16, 0, HD, lane);
            v16h b = ldsFragRowK(lds_k, nt * 16, 0, HD, lane);  // B[k][n] = K[n][k]
            v8f c = {};
            c = WMMA_F16(a, b, c);
            int n = nt * 16 + (lane & 15), mh = (lane >> 4) << 3;
#pragma unroll
            for (int r = 0; r < 8; ++r) {
                int m = mt * 16 + mh + r;
                float v = c[r];
                if (m >= 2 && m < N && n >= 2 && n < N) {
                    int i = m - 2, j = n - 2;
                    int idx = (((i >> 3) - (j >> 3) + 7) * 15) + ((i & 7) - (j & 7) + 7);
                    v += bias_table[idx * H + h];
                }
                lds_af[m * NP + n] = v;
            }
        }
        __syncthreads();

        // ---- Phase D: softmax. Reductions 1 thread/row; exp pass 256-wide ----
        if (tid < NP) {
            if (tid < N) {
                const float* row = lds_af + tid * NP;
                float mx = -3.0e38f;
                for (int j = 0; j < N; ++j) mx = fmaxf(mx, row[j]);
                float s = 0.f;
                for (int j = 0; j < N; ++j) s += __expf(row[j] - mx);
                row_mx[tid]  = mx;
                row_inv[tid] = 1.f / s;
            } else {
                row_mx[tid]  = 0.f;
                row_inv[tid] = 0.f;
            }
        }
        __syncthreads();
        for (int i = tid; i < NP * APW; i += 256) {
            int m = i / APW, j = i % APW;
            float v = (m < N && j < N)
                          ? __expf(lds_af[m * NP + j] - row_mx[m]) * row_inv[m]
                          : 0.f;
            lds_ah[i] = (_Float16)v;
        }
        __syncthreads();

        // ---- Phase E: out_h = attn @ v (10 tiles, K=96 in 3 steps) ----
        for (int t = wave; t < 10; t += 8) {
            int mt = t >> 1, nt = t & 1;
            v8f c = {};
#pragma unroll
            for (int ks = 0; ks < 3; ++ks) {
                v16h a = ldsFragRowK(lds_ah, mt * 16, ks * 32, APW, lane);
                v16h b = ldsFragRowK(lds_vt, nt * 16, ks * 32, APW, lane);
                c = WMMA_F16(a, b, c);
            }
            int n = lane & 15, mh = (lane >> 4) << 3;
#pragma unroll
            for (int r = 0; r < 8; ++r) {
                int m = mt * 16 + mh + r;
                lds_o[m * C + h * HD + nt * 16 + n] = (_Float16)c[r];
            }
        }
        __syncthreads();   // protects q/k/vt reuse next head iteration
    }

    // ---- Phase F: out = o @ proj_w^T + proj_b (80 tiles) ----
    const _Float16* ppk = wpk + (size_t)QKV_FRAGS * 32 * 16;
    float* ow = out + (size_t)win * N * C;
    for (int t = wave; t < 80; t += 8) {
        int mt = t >> 4, nt = t & 15;
        v8f c = {};
#pragma unroll
        for (int ks = 0; ks < NKSTEPS; ++ks) {
            v16h a = ldsFragRowK(lds_o, mt * 16, ks * 32, C, lane);
            v16h b = pkFrag(ppk, nt * NKSTEPS + ks, lane);
            c = WMMA_F16(a, b, c);
        }
        int n = nt * 16 + (lane & 15), mh = (lane >> 4) << 3;
        float pb = proj_b[n];
#pragma unroll
        for (int r = 0; r < 8; ++r) {
            int m = mt * 16 + mh + r;
            if (m < N) ow[(size_t)m * C + n] = c[r] + pb;
        }
    }
}

extern "C" void kernel_launch(void* const* d_in, const int* in_sizes, int n_in,
                              void* d_out, int out_size, void* d_ws, size_t ws_size,
                              hipStream_t stream) {
    const float* x          = (const float*)d_in[0];
    const float* qkv_w      = (const float*)d_in[1];
    const float* qkv_b      = (const float*)d_in[2];
    const float* bias_table = (const float*)d_in[3];
    const float* proj_w     = (const float*)d_in[4];
    const float* proj_b     = (const float*)d_in[5];
    float*       out        = (float*)d_out;
    _Float16*    wpk        = (_Float16*)d_ws;   // 512 KB packed f16 weights

    const int n_tok = 66, dim = 256;
    const int n_win = in_sizes[0] / (n_tok * dim);   // 2048

    // Pre-pack weights into WMMA B-fragment layout (f16) in workspace.
    const int pack_threads = TOTAL_FRAGS * 32;       // 16384
    pack_weights_kernel<<<(pack_threads + 255) / 256, 256, 0, stream>>>(qkv_w, proj_w, wpk);

    winattn_fused_kernel<<<n_win, 256, 0, stream>>>(
        x, wpk, qkv_b, bias_table, proj_b, out);
}
